// MultiHeadAttention_21431886807341
// MI455X (gfx1250) — compile-verified, matrix operand order checked by reading, not run on hardware
//
#include <hip/hip_runtime.h>

// ---------------------------------------------------------------- types
typedef __attribute__((ext_vector_type(16))) __bf16 v16bf;
typedef __attribute__((ext_vector_type(8)))  __bf16 v8bf;
typedef __attribute__((ext_vector_type(8)))  float  v8f;
typedef __attribute__((ext_vector_type(4)))  int    vi4;

#define NB   4
#define SEQ  2048
#define DM   1024
#define NH   16
#define DK   64

#define AS_GLOBAL __attribute__((address_space(1)))
#define AS_LDS    __attribute__((address_space(3)))

__device__ __forceinline__ v8f zero8() {
    v8f z;
#pragma unroll
    for (int i = 0; i < 8; ++i) z[i] = 0.0f;
    return z;
}

__device__ __forceinline__ v16bf cat16(v8bf lo, v8bf hi) {
    v16bf r;
#pragma unroll
    for (int i = 0; i < 8; ++i) { r[i] = lo[i]; r[8 + i] = hi[i]; }
    return r;
}

// A/B fragment (16x32 bf16) per ISA layout: lane m = lane&15,
// kb = (lane>=16)?8:0, elems 0..7 = K[kb..kb+7], 8..15 = K[kb+16..kb+23].
__device__ __forceinline__ v16bf frag_bf16(const __bf16* p) {
    v8bf lo = *reinterpret_cast<const v8bf*>(p);
    v8bf hi = *reinterpret_cast<const v8bf*>(p + 16);
    return cat16(lo, hi);
}

// Same fragment but converting fp32 source inline (two float4 pairs).
__device__ __forceinline__ v16bf frag_f32(const float* p) {
    const float4* p4 = reinterpret_cast<const float4*>(p);
    float4 f0 = p4[0], f1 = p4[1], f2 = p4[4], f3 = p4[5];
    v16bf r;
    r[0] = (__bf16)f0.x; r[1]  = (__bf16)f0.y; r[2]  = (__bf16)f0.z; r[3]  = (__bf16)f0.w;
    r[4] = (__bf16)f1.x; r[5]  = (__bf16)f1.y; r[6]  = (__bf16)f1.z; r[7]  = (__bf16)f1.w;
    r[8] = (__bf16)f2.x; r[9]  = (__bf16)f2.y; r[10] = (__bf16)f2.z; r[11] = (__bf16)f2.w;
    r[12] = (__bf16)f3.x; r[13] = (__bf16)f3.y; r[14] = (__bf16)f3.z; r[15] = (__bf16)f3.w;
    return r;
}

__device__ __forceinline__ v8f wmma_bf16(v16bf a, v16bf b, v8f c) {
    return __builtin_amdgcn_wmma_f32_16x16x32_bf16(false, a, false, b, (short)0, c, false, false);
}

// ---- async global->LDS staging (CDNA5 ASYNCcnt path), guarded fallback ----
__device__ __forceinline__ void stage_async_b128(const void* g, void* l) {
#if __has_builtin(__builtin_amdgcn_global_load_async_to_lds_b128)
    __builtin_amdgcn_global_load_async_to_lds_b128(
        (AS_GLOBAL vi4*)(g), (AS_LDS vi4*)(l), 0, 0);
#else
    *(float4*)l = *(const float4*)g;
#endif
}

__device__ __forceinline__ void stage_wait() {
#if __has_builtin(__builtin_amdgcn_global_load_async_to_lds_b128)
#if __has_builtin(__builtin_amdgcn_s_wait_asynccnt)
    __builtin_amdgcn_s_wait_asynccnt(0);
#else
    asm volatile("s_wait_asynccnt 0" ::: "memory");
#endif
#endif
    __syncthreads();
}

// ---------------------------------------------------------------- kernel 1:
// transpose-convert the 4 weight matrices to bf16 W^T (row = out-col, contiguous K)
__global__ __launch_bounds__(256) void wconv_kernel(const float* __restrict__ Wq,
                                                    const float* __restrict__ Wk,
                                                    const float* __restrict__ Wv,
                                                    const float* __restrict__ Wo,
                                                    __bf16* __restrict__ Wt) {
    int w = blockIdx.y;
    const float* src = (w == 0) ? Wq : (w == 1) ? Wk : (w == 2) ? Wv : Wo;
    __bf16* dst = Wt + (size_t)w * DM * DM;
    int idx = blockIdx.x * 256 + threadIdx.x;   // n*1024 + k
    int n = idx >> 10, k = idx & (DM - 1);
    dst[idx] = (__bf16)src[(size_t)k * DM + n];
}

// ---------------------------------------------------------------- kernel 2:
// Y = X @ W + b, output in per-head layout (bf16). transposed=1 stores [b,h,dk,s].
// The 16-row fp32 A tile is staged once per block into LDS (async), shared by
// all 8 waves; each wave owns one 16-col tile of W^T.
__global__ __launch_bounds__(256) void proj_kernel(const float* __restrict__ X,
                                                   const __bf16* __restrict__ Wt,
                                                   const float* __restrict__ bias,
                                                   __bf16* __restrict__ out,
                                                   int transposed) {
    __shared__ __attribute__((aligned(16))) float As[16][DM];   // 64 KB

    int lane = threadIdx.x & 31, wave = threadIdx.x >> 5;
    int m0 = blockIdx.x * 16;
    int n0 = (blockIdx.y * 8 + wave) * 16;
    int colB = n0 + (lane & 15);
    int kb   = (lane >> 4) * 8;

    // stage A tile: 16 rows x 1024 f32 = 64 KB, 16B per lane per shot
    {
        const float* gsrc = X + (size_t)m0 * DM;
        float* lbase = &As[0][0];
#pragma unroll
        for (int it = 0; it < 16; ++it) {
            int idx = (it * 256 + threadIdx.x) * 4;
            stage_async_b128(gsrc + idx, lbase + idx);
        }
    }
    stage_wait();

    const float*  ap = &As[lane & 15][kb];
    const __bf16* bp = Wt + (size_t)colB * DM + kb;

    v8f c = zero8();
#pragma unroll 4
    for (int k0 = 0; k0 < DM; k0 += 32) {
        v16bf a = frag_f32(ap + k0);
        v16bf b = frag_bf16(bp + k0);
        c = wmma_bf16(a, b, c);
    }

    float bv = bias[colB];
    int h = colB >> 6, d = colB & (DK - 1);
#pragma unroll
    for (int i = 0; i < 8; ++i) {
        int M = i + ((lane >> 4) << 3);
        int r = m0 + M;
        int bidx = r >> 11, s = r & (SEQ - 1);
        float val = c[i] + bv;
        if (transposed)
            out[(((size_t)(bidx * NH + h)) * DK + d) * SEQ + s] = (__bf16)val;
        else
            out[(((size_t)(bidx * NH + h)) * SEQ + s) * DK + d] = (__bf16)val;
    }
}

// ---------------------------------------------------------------- kernel 3:
// flash attention: one 16-row Q tile per wave, 32 keys per iteration.
__global__ __launch_bounds__(256) void attn_kernel(const __bf16* __restrict__ Qh,
                                                   const __bf16* __restrict__ Kh,
                                                   const __bf16* __restrict__ Vt,
                                                   const int*    __restrict__ mask,
                                                   __bf16* __restrict__ ctx) {
    __shared__ __attribute__((aligned(16))) __bf16 pbuf[8][16][32];

    int lane = threadIdx.x & 31, wave = threadIdx.x >> 5;
    int bh = blockIdx.y;                 // b*NH + h
    int b  = bh >> 4, h = bh & (NH - 1);
    int q0 = (blockIdx.x * 8 + wave) * 16;
    int N  = lane & 15;
    int kb = (lane >> 4) * 8;

    // resident Q fragments (dk = 64 -> two 16x32 fragments)
    const __bf16* qbase = Qh + ((size_t)bh * SEQ + q0 + N) * DK;
    v16bf qa0 = frag_bf16(qbase + kb);
    v16bf qa1 = frag_bf16(qbase + 32 + kb);

    v8f o[4];
#pragma unroll
    for (int f = 0; f < 4; ++f) o[f] = zero8();
    float mrow[8], lrow[8];
#pragma unroll
    for (int i = 0; i < 8; ++i) { mrow[i] = -1e30f; lrow[i] = 0.0f; }

    const int* mbase = mask + (size_t)b * SEQ * SEQ;

    for (int s0 = 0; s0 < SEQ; s0 += 32) {
        // --- scores: two 16x16 tiles (keys s0..+15, s0+16..+31)
        const __bf16* kB0 = Kh + ((size_t)bh * SEQ + s0 + N) * DK;
        const __bf16* kB1 = kB0 + 16 * DK;
        v8f st0 = wmma_bf16(qa1, frag_bf16(kB0 + 32 + kb),
                  wmma_bf16(qa0, frag_bf16(kB0 + kb), zero8()));
        v8f st1 = wmma_bf16(qa1, frag_bf16(kB1 + 32 + kb),
                  wmma_bf16(qa0, frag_bf16(kB1 + kb), zero8()));

        if (s0 + 32 < SEQ)
            __builtin_prefetch(kB0 + 32 * DK, 0, 1);

        // --- scale, mask, online softmax (rows live across 16-lane halves)
#pragma unroll
        for (int i = 0; i < 8; ++i) {
            int M = i + ((lane >> 4) << 3);
            size_t roff = (size_t)(q0 + M) * SEQ + s0;
            float v0 = st0[i] * 0.125f;
            float v1 = st1[i] * 0.125f;
            if (mbase[roff + N] == 0)      v0 = -1e9f;
            if (mbase[roff + 16 + N] == 0) v1 = -1e9f;

            float t = fmaxf(v0, v1);
            t = fmaxf(t, __shfl_xor(t, 1, 16));
            t = fmaxf(t, __shfl_xor(t, 2, 16));
            t = fmaxf(t, __shfl_xor(t, 4, 16));
            t = fmaxf(t, __shfl_xor(t, 8, 16));
            float mnew  = fmaxf(mrow[i], t);
            float alpha = __expf(mrow[i] - mnew);
            float p0 = __expf(v0 - mnew);
            float p1 = __expf(v1 - mnew);
            float rs = p0 + p1;
            rs += __shfl_xor(rs, 1, 16);
            rs += __shfl_xor(rs, 2, 16);
            rs += __shfl_xor(rs, 4, 16);
            rs += __shfl_xor(rs, 8, 16);
            lrow[i] = lrow[i] * alpha + rs;
            mrow[i] = mnew;
            o[0][i] *= alpha; o[1][i] *= alpha; o[2][i] *= alpha; o[3][i] *= alpha;

            pbuf[wave][M][N]      = (__bf16)p0;
            pbuf[wave][M][16 + N] = (__bf16)p1;
        }

        asm volatile("s_wait_dscnt 0" ::: "memory");

        // --- P (16x32) as A fragment from this wave's private LDS tile
        v16bf pa = frag_bf16(&pbuf[wave][lane & 15][kb]);

        // --- O += P @ V  (V^T gives contiguous B-fragment columns)
#pragma unroll
        for (int f = 0; f < 4; ++f) {
            int dcol = f * 16 + N;
            const __bf16* vb = Vt + ((size_t)bh * DK + dcol) * SEQ + s0 + kb;
            o[f] = wmma_bf16(pa, frag_bf16(vb), o[f]);
        }
    }

    // --- normalize and store ctx (bf16, [B,S,D] with D = h*64+d)
#pragma unroll
    for (int i = 0; i < 8; ++i) {
        int M = i + ((lane >> 4) << 3);
        float inv = 1.0f / lrow[i];
        size_t base = ((size_t)(b * SEQ + q0 + M)) * DM + h * DK + N;
        ctx[base +  0] = (__bf16)(o[0][i] * inv);
        ctx[base + 16] = (__bf16)(o[1][i] * inv);
        ctx[base + 32] = (__bf16)(o[2][i] * inv);
        ctx[base + 48] = (__bf16)(o[3][i] * inv);
    }
}

// ---------------------------------------------------------------- kernel 4:
// out = LN(ctx @ Wo + bo + residual); one block = 16 full rows.
// The shared bf16 ctx A tile (32 KB) is staged async into LDS.
__global__ __launch_bounds__(256) void oproj_ln_kernel(const __bf16* __restrict__ ctx,
                                                       const __bf16* __restrict__ Wot,
                                                       const float* __restrict__ bo,
                                                       const float* __restrict__ residual,
                                                       const float* __restrict__ gamma,
                                                       const float* __restrict__ beta,
                                                       float* __restrict__ out) {
    __shared__ __attribute__((aligned(16))) __bf16 Abuf[16][DM];  // 32 KB
    __shared__ __attribute__((aligned(16))) float  xrow[16][DM];  // 64 KB

    int lane = threadIdx.x & 31, wave = threadIdx.x >> 5;
    int m0 = blockIdx.x * 16;
    int kb = (lane >> 4) * 8;

    // stage A tile: 16 rows x 1024 bf16 = 32 KB
    {
        const __bf16* gsrc = ctx + (size_t)m0 * DM;
        __bf16* lbase = &Abuf[0][0];
#pragma unroll
        for (int it = 0; it < 8; ++it) {
            int idx = (it * 256 + threadIdx.x) * 8;
            stage_async_b128(gsrc + idx, lbase + idx);
        }
    }
    stage_wait();

    const __bf16* ap = &Abuf[lane & 15][kb];

#pragma unroll
    for (int t = 0; t < 8; ++t) {
        int colB = (wave * 8 + t) * 16 + (lane & 15);
        const __bf16* bp = Wot + (size_t)colB * DM + kb;
        v8f c = zero8();
#pragma unroll 4
        for (int k0 = 0; k0 < DM; k0 += 32)
            c = wmma_bf16(frag_bf16(ap + k0), frag_bf16(bp + k0), c);
        float bv = bo[colB];
#pragma unroll
        for (int i = 0; i < 8; ++i) {
            int M = i + ((lane >> 4) << 3);
            float val = c[i] + bv + residual[(size_t)(m0 + M) * DM + colB];
            xrow[M][colB] = val;
        }
    }

    __syncthreads();

    // LayerNorm: 8 waves x 2 rows
#pragma unroll
    for (int rr = 0; rr < 2; ++rr) {
        int r = wave * 2 + rr;
        float sum = 0.0f, sq = 0.0f;
#pragma unroll
        for (int j = 0; j < DM / 32; ++j) {
            float x = xrow[r][lane + 32 * j];
            sum += x; sq += x * x;
        }
#pragma unroll
        for (int d = 1; d < 32; d <<= 1) {
            sum += __shfl_xor(sum, d, 32);
            sq  += __shfl_xor(sq,  d, 32);
        }
        float mu   = sum * (1.0f / DM);
        float var  = sq * (1.0f / DM) - mu * mu;
        float rinv = rsqrtf(var + 1e-6f);
        size_t grow = (size_t)(m0 + r) * DM;
#pragma unroll
        for (int j = 0; j < DM / 32; ++j) {
            int col = lane + 32 * j;
            float x = xrow[r][col];
            out[grow + col] = (x - mu) * rinv * gamma[col] + beta[col];
        }
    }
}

// ---------------------------------------------------------------- launcher
extern "C" void kernel_launch(void* const* d_in, const int* in_sizes, int n_in,
                              void* d_out, int out_size, void* d_ws, size_t ws_size,
                              hipStream_t stream) {
    const float* q    = (const float*)d_in[0];
    const float* k    = (const float*)d_in[1];
    const float* v    = (const float*)d_in[2];
    const int*   mask = (const int*)  d_in[3];
    const float* Wq   = (const float*)d_in[4];
    const float* bq   = (const float*)d_in[5];
    const float* Wk   = (const float*)d_in[6];
    const float* bk   = (const float*)d_in[7];
    const float* Wv   = (const float*)d_in[8];
    const float* bv   = (const float*)d_in[9];
    const float* Wo   = (const float*)d_in[10];
    const float* bo   = (const float*)d_in[11];
    const float* gamma = (const float*)d_in[12];
    const float* beta  = (const float*)d_in[13];
    float* out = (float*)d_out;

    char* ws = (char*)d_ws;
    const size_t WSZ  = (size_t)DM * DM;              // per-weight elems
    const size_t HSZ  = (size_t)NB * NH * SEQ * DK;   // per-head-tensor elems
    __bf16* Wt  = (__bf16*)ws;                                        // 4*WSZ bf16
    __bf16* Qh  = (__bf16*)(ws + 4 * WSZ * 2);                        // HSZ
    __bf16* Kh  = Qh + HSZ;
    __bf16* Vt  = Kh + HSZ;
    __bf16* CTX = Vt + HSZ;                                           // HSZ

    wconv_kernel<<<dim3(WSZ / 256, 4), 256, 0, stream>>>(Wq, Wk, Wv, Wo, Wt);

    dim3 pg(NB * SEQ / 16, DM / (16 * 8));
    proj_kernel<<<pg, 256, 0, stream>>>(q, Wt,           bq, Qh, 0);
    proj_kernel<<<pg, 256, 0, stream>>>(k, Wt + WSZ,     bk, Kh, 0);
    proj_kernel<<<pg, 256, 0, stream>>>(v, Wt + 2 * WSZ, bv, Vt, 1);

    attn_kernel<<<dim3(SEQ / 128, NB * NH), 256, 0, stream>>>(Qh, Kh, Vt, mask, CTX);

    oproj_ln_kernel<<<dim3(NB * SEQ / 16), 256, 0, stream>>>(
        CTX, Wt + 3 * WSZ, bo, q, gamma, beta, out);
}
